// TriangleAttentionCore_31937376813596
// MI455X (gfx1250) — matrix operand order repaired and checked by reading, not compiled, not run on hardware
//
#include <hip/hip_runtime.h>

typedef __attribute__((ext_vector_type(16))) __bf16 v16bf;
typedef __attribute__((ext_vector_type(8)))  __bf16 v8bf;
typedef __attribute__((ext_vector_type(8)))  float  v8f;

#define NSEQ   256
#define CZ     128
#define NHEAD  4
#define CHD    32
#define DMODEL 128            // NHEAD*CHD
#define NROWS  (NSEQ*NSEQ)    // 65536
#define ATT_SCALE 0.17677669529663687f  // 1/sqrt(32)

// ---------- bf16 helpers (bit-exact RNE, no reliance on __bf16 conversions) ----------
__device__ __forceinline__ __bf16 f2bf(float f) {
    union { float f; unsigned u; } x; x.f = f;
    unsigned r = (x.u + 0x7FFFu + ((x.u >> 16) & 1u)) >> 16;
    union { unsigned short s; __bf16 h; } y; y.s = (unsigned short)r;
    return y.h;
}
__device__ __forceinline__ float bf2f(__bf16 h) {
    union { unsigned short s; __bf16 h; } y; y.h = h;
    union { unsigned u; float f; } x; x.u = ((unsigned)y.s) << 16;
    return x.f;
}
__device__ __forceinline__ v16bf cat16(v8bf lo, v8bf hi) {
    v16bf r;
#pragma unroll
    for (int t = 0; t < 8; ++t) { r[t] = lo[t]; r[t + 8] = hi[t]; }
    return r;
}

// ---------- K0: convert + transpose weights: Wt[w][n*128+k] = (bf16)W[w][k*128+n] ----------
__global__ void k_prep_w(const float* Wq, const float* Wk, const float* Wv,
                         const float* Wg, const float* Wo, __bf16* wt) {
    const float* src;
    switch (blockIdx.y) {
        case 0: src = Wq; break;
        case 1: src = Wk; break;
        case 2: src = Wv; break;
        case 3: src = Wg; break;
        default: src = Wo; break;
    }
    int idx = blockIdx.x * blockDim.x + threadIdx.x;   // 0..16383  (= k*128 + n)
    int kk = idx >> 7, nn = idx & 127;
    wt[blockIdx.y * 16384 + nn * 128 + kk] = f2bf(src[idx]);
}

// ---------- K1: LayerNorm over CZ; one wave32 per row ----------
__global__ void k_layernorm(const float* __restrict__ z, const float* __restrict__ lnw,
                            const float* __restrict__ lnb, __bf16* __restrict__ zn) {
    int wv = threadIdx.x >> 5, lane = threadIdx.x & 31;
    int row = blockIdx.x * 8 + wv;
    const float4 x4 = reinterpret_cast<const float4*>(z + (size_t)row * CZ)[lane];
    float xs[4] = {x4.x, x4.y, x4.z, x4.w};
    float s = xs[0] + xs[1] + xs[2] + xs[3];
    float q = xs[0]*xs[0] + xs[1]*xs[1] + xs[2]*xs[2] + xs[3]*xs[3];
#pragma unroll
    for (int m = 16; m >= 1; m >>= 1) {
        s += __shfl_xor(s, m, 32);
        q += __shfl_xor(q, m, 32);
    }
    float mean = s * (1.0f / CZ);
    float var  = q * (1.0f / CZ) - mean * mean;
    float inv  = rsqrtf(var + 1e-5f);
#pragma unroll
    for (int c = 0; c < 4; ++c) {
        int col = lane * 4 + c;
        float v = (xs[c] - mean) * inv * lnw[col] + lnb[col];
        zn[(size_t)row * CZ + col] = f2bf(v);
    }
}

// ---------- K2: bf16 WMMA GEMM  A[M,128] x Bt[n*128+k] -> 16x16 tile per wave ----------
// MODE 0: bf16 out[row*128+n]             (g projection)
// MODE 1: bf16 out[((i*4+h)*256+j)*32+ch] (q,k: [i,h,j,ch])
// MODE 2: bf16 out[((i*4+h)*32+ch)*256+j] (vT:  [i,h,ch,j])
// MODE 3: f32  out[row*128+n]             (final output)
template <int MODE>
__global__ void k_gemm(const __bf16* __restrict__ A, const __bf16* __restrict__ Bt,
                       void* __restrict__ out) {
    int wv = threadIdx.x >> 5, lane = threadIdx.x & 31;
    int lm = lane & 15, half = lane >> 4;
    int rowBase = blockIdx.x * 16;
    int nBase = wv * 16;

    v8f acc = {0.f, 0.f, 0.f, 0.f, 0.f, 0.f, 0.f, 0.f};
    const __bf16* arow = A + (size_t)(rowBase + lm) * 128;
    const __bf16* brow = Bt + (size_t)(nBase + lm) * 128;
#pragma unroll
    for (int kb = 0; kb < 4; ++kb) {
        v8bf alo = *reinterpret_cast<const v8bf*>(arow + kb * 32 + half * 8);
        v8bf ahi = *reinterpret_cast<const v8bf*>(arow + kb * 32 + half * 8 + 16);
        v16bf a = cat16(alo, ahi);
        v8bf blo = *reinterpret_cast<const v8bf*>(brow + kb * 32 + half * 8);
        v8bf bhi = *reinterpret_cast<const v8bf*>(brow + kb * 32 + half * 8 + 16);
        v16bf b = cat16(blo, bhi);
        acc = __builtin_amdgcn_wmma_f32_16x16x32_bf16(false, a, false, b,
                                                      (short)0, acc, false, false);
    }

    int n = nBase + lm;
#pragma unroll
    for (int r = 0; r < 8; ++r) {
        int R = rowBase + r + 8 * half;
        if (MODE == 3) {
            reinterpret_cast<float*>(out)[(size_t)R * 128 + n] = acc[r];
        } else if (MODE == 0) {
            reinterpret_cast<__bf16*>(out)[(size_t)R * 128 + n] = f2bf(acc[r]);
        } else {
            int i = R >> 8, j = R & 255, h = n >> 5, ch = n & 31;
            size_t idx = (MODE == 1)
                ? ((size_t)(i * 4 + h) * 256 + j) * 32 + ch
                : ((size_t)(i * 4 + h) * 32 + ch) * 256 + j;
            reinterpret_cast<__bf16*>(out)[idx] = f2bf(acc[r]);
        }
    }
}

// ---------- K3: flash-style attention per (i,h) ----------
// Online softmax: only one 32-key logits slab live at a time -> no VGPR spills.
__global__ void k_attn(const __bf16* __restrict__ q, const __bf16* __restrict__ k,
                       const __bf16* __restrict__ vT, const float* __restrict__ mask_bias,
                       const float* __restrict__ edge_bias, float* __restrict__ o_ws) {
    __shared__ __bf16 P[4][16][32];  // per-wave 16x32 unnormalized probs (1 KB each)
    int wv = threadIdx.x >> 5, lane = threadIdx.x & 31;
    int lm = lane & 15, half = lane >> 4;
    int ih = blockIdx.x, i = ih >> 2, h = ih & 3;
    const __bf16* qih = q  + (size_t)ih * NSEQ * CHD;
    const __bf16* kih = k  + (size_t)ih * NSEQ * CHD;
    const __bf16* vih = vT + (size_t)ih * CHD * NSEQ;
    const float*  mb  = mask_bias + (size_t)i * NSEQ;
    const float*  eb  = edge_bias + (size_t)h * NSEQ * NSEQ;

    for (int s = 0; s < 4; ++s) {
        int qt = wv * 4 + s;
        // A = q tile [16 x 32]
        const __bf16* qr = qih + (size_t)(qt * 16 + lm) * CHD + half * 8;
        v16bf aq = cat16(*reinterpret_cast<const v8bf*>(qr),
                         *reinterpret_cast<const v8bf*>(qr + 16));

        v8f o0 = {0.f, 0.f, 0.f, 0.f, 0.f, 0.f, 0.f, 0.f};
        v8f o1 = o0;
        float m_run[8], l_run[8];
#pragma unroll
        for (int r = 0; r < 8; ++r) { m_run[r] = -3.4e38f; l_run[r] = 0.f; }

        for (int kb = 0; kb < 8; ++kb) {   // 32 keys per step
            // logits for two 16-key sub-tiles (K = CHD = 32 -> one WMMA each)
            const __bf16* kr0 = kih + (size_t)(kb * 32 + lm) * CHD + half * 8;
            const __bf16* kr1 = kih + (size_t)(kb * 32 + 16 + lm) * CHD + half * 8;
            v16bf bk0 = cat16(*reinterpret_cast<const v8bf*>(kr0),
                              *reinterpret_cast<const v8bf*>(kr0 + 16));
            v16bf bk1 = cat16(*reinterpret_cast<const v8bf*>(kr1),
                              *reinterpret_cast<const v8bf*>(kr1 + 16));
            v8f zc = {0.f, 0.f, 0.f, 0.f, 0.f, 0.f, 0.f, 0.f};
            v8f acc0 = __builtin_amdgcn_wmma_f32_16x16x32_bf16(false, aq, false, bk0,
                                                               (short)0, zc, false, false);
            v8f acc1 = __builtin_amdgcn_wmma_f32_16x16x32_bf16(false, aq, false, bk1,
                                                               (short)0, zc, false, false);
            // biases
            float mb0 = mb[kb * 32 + lm];
            float mb1 = mb[kb * 32 + 16 + lm];
            float tl0[8], tl1[8], tmax[8];
#pragma unroll
            for (int r = 0; r < 8; ++r) {
                int qrow = qt * 16 + r + 8 * half;
                const float* ebr = eb + (size_t)qrow * NSEQ + kb * 32;
                tl0[r] = acc0[r] * ATT_SCALE + mb0 + ebr[lm];
                tl1[r] = acc1[r] * ATT_SCALE + mb1 + ebr[16 + lm];
                tmax[r] = fmaxf(tl0[r], tl1[r]);
            }
#pragma unroll
            for (int r = 0; r < 8; ++r) {
#pragma unroll
                for (int m = 1; m <= 8; m <<= 1)
                    tmax[r] = fmaxf(tmax[r], __shfl_xor(tmax[r], m, 16));
            }
            // online-softmax update; write unnormalized probs to LDS
            float alpha[8], tsum[8];
#pragma unroll
            for (int r = 0; r < 8; ++r) {
                float mn = fmaxf(m_run[r], tmax[r]);
                alpha[r] = __expf(m_run[r] - mn);
                m_run[r] = mn;
                float p0 = __expf(tl0[r] - mn);
                float p1 = __expf(tl1[r] - mn);
                int Mrow = r + 8 * half;
                P[wv][Mrow][lm]      = f2bf(p0);
                P[wv][Mrow][16 + lm] = f2bf(p1);
                tsum[r] = p0 + p1;
            }
#pragma unroll
            for (int r = 0; r < 8; ++r) {
#pragma unroll
                for (int m = 1; m <= 8; m <<= 1)
                    tsum[r] += __shfl_xor(tsum[r], m, 16);
            }
#pragma unroll
            for (int r = 0; r < 8; ++r) {
                l_run[r] = l_run[r] * alpha[r] + tsum[r];
                o0[r] *= alpha[r];
                o1[r] *= alpha[r];
            }
            // P @ V for this 32-key slab: A-layout reload from LDS, B = vT rows
            const __bf16* pr = &P[wv][lm][half * 8];
            v16bf ap = cat16(*reinterpret_cast<const v8bf*>(pr),
                             *reinterpret_cast<const v8bf*>(pr + 16));
            const __bf16* v0 = vih + (size_t)lm * NSEQ + kb * 32 + half * 8;
            const __bf16* v1 = vih + (size_t)(16 + lm) * NSEQ + kb * 32 + half * 8;
            v16bf b0 = cat16(*reinterpret_cast<const v8bf*>(v0),
                             *reinterpret_cast<const v8bf*>(v0 + 16));
            v16bf b1 = cat16(*reinterpret_cast<const v8bf*>(v1),
                             *reinterpret_cast<const v8bf*>(v1 + 16));
            o0 = __builtin_amdgcn_wmma_f32_16x16x32_bf16(false, ap, false, b0,
                                                         (short)0, o0, false, false);
            o1 = __builtin_amdgcn_wmma_f32_16x16x32_bf16(false, ap, false, b1,
                                                         (short)0, o1, false, false);
        }
        // normalize and store pre-gate o: [row = i*256+qrow][h*32+ch] fp32
#pragma unroll
        for (int r = 0; r < 8; ++r) {
            float inv = 1.0f / l_run[r];
            int qrow = qt * 16 + r + 8 * half;
            size_t rowoff = ((size_t)i * NSEQ + qrow) * DMODEL + h * CHD;
            o_ws[rowoff + lm]      = o0[r] * inv;
            o_ws[rowoff + 16 + lm] = o1[r] * inv;
        }
    }
}

// ---------- K4: gated = o * sigmoid(g), bf16 out ----------
__global__ void k_gate(const float* __restrict__ o_ws, const __bf16* __restrict__ g,
                       __bf16* __restrict__ gated) {
    size_t idx = (size_t)blockIdx.x * blockDim.x + threadIdx.x;
    float gv = bf2f(g[idx]);
    float sg = 1.0f / (1.0f + __expf(-gv));
    gated[idx] = f2bf(o_ws[idx] * sg);
}

extern "C" void kernel_launch(void* const* d_in, const int* in_sizes, int n_in,
                              void* d_out, int out_size, void* d_ws, size_t ws_size,
                              hipStream_t stream) {
    (void)in_sizes; (void)n_in; (void)out_size; (void)ws_size;
    const float* z    = (const float*)d_in[0];
    const float* mb   = (const float*)d_in[1];
    const float* eb   = (const float*)d_in[2];
    const float* lnw  = (const float*)d_in[3];
    const float* lnb  = (const float*)d_in[4];
    const float* Wq   = (const float*)d_in[5];
    const float* Wk   = (const float*)d_in[6];
    const float* Wv   = (const float*)d_in[7];
    const float* Wg   = (const float*)d_in[8];
    const float* Wo   = (const float*)d_in[9];

    char* ws = (char*)d_ws;
    const size_t SZ_BF = (size_t)NROWS * DMODEL * sizeof(__bf16);   // 16.78 MB
    __bf16* wt    = (__bf16*)(ws);                                  // 5*16384 bf16
    __bf16* zn    = (__bf16*)(ws + 262144);
    __bf16* qb    = (__bf16*)(ws + 262144 + 1 * SZ_BF);
    __bf16* kb    = (__bf16*)(ws + 262144 + 2 * SZ_BF);
    __bf16* vTb   = (__bf16*)(ws + 262144 + 3 * SZ_BF);
    __bf16* gb    = (__bf16*)(ws + 262144 + 4 * SZ_BF);
    float*  o_ws  = (float*) (ws + 262144 + 5 * SZ_BF);
    __bf16* gated = zn;  // reuse zn region after projections are done

    __bf16* WtQ = wt;           __bf16* WtK = wt + 16384;
    __bf16* WtV = wt + 2*16384; __bf16* WtG = wt + 3*16384;
    __bf16* WtO = wt + 4*16384;

    k_prep_w<<<dim3(64, 5), 256, 0, stream>>>(Wq, Wk, Wv, Wg, Wo, wt);
    k_layernorm<<<NROWS / 8, 256, 0, stream>>>(z, lnw, lnb, zn);

    k_gemm<1><<<NROWS / 16, 256, 0, stream>>>(zn, WtQ, qb);
    k_gemm<1><<<NROWS / 16, 256, 0, stream>>>(zn, WtK, kb);
    k_gemm<2><<<NROWS / 16, 256, 0, stream>>>(zn, WtV, vTb);
    k_gemm<0><<<NROWS / 16, 256, 0, stream>>>(zn, WtG, gb);

    k_attn<<<NSEQ * NHEAD, 128, 0, stream>>>(qb, kb, vTb, mb, eb, o_ws);

    k_gate<<<(NROWS * DMODEL) / 256, 256, 0, stream>>>(o_ws, gb, gated);

    k_gemm<3><<<NROWS / 16, 256, 0, stream>>>(gated, WtO, d_out);
}
